// HierarchicalTimeSeriesModel_28810640621860
// MI455X (gfx1250) — matrix operand order repaired and checked by reading, not compiled, not run on hardware
//
#include <hip/hip_runtime.h>
#include <hip/hip_bf16.h>

typedef __attribute__((ext_vector_type(16))) _Float16 v16h;
typedef __attribute__((ext_vector_type(8)))  _Float16 v8h;
typedef __attribute__((ext_vector_type(8)))  float    v8f;

#define B_   2048
#define T_   1024
#define IN_  5
#define H_   64
#define TT_  32     // timesteps staged in LDS per chunk

#define LOG2E_ 1.4426950408889634f
#define LN2_   0.6931471805599453f

// Branch-free tanh via raw v_exp_f32 (base-2) + v_rcp_f32:
// tanh(x) = 1 - 2/(e^{2x}+1); saturates correctly at +-1 for |x| large.
__device__ __forceinline__ float fast_tanh(float x) {
    float e = __builtin_amdgcn_exp2f(2.0f * LOG2E_ * x);
    return 1.0f - 2.0f * __builtin_amdgcn_rcpf(e + 1.0f);
}
// Branch-free softplus via raw v_exp_f32 / v_log_f32 (both base-2):
// softplus(x) = ln2 * log2(1 + 2^{x*log2e}); cndmask guard for overflow.
__device__ __forceinline__ float softplus_f(float x) {
    float e  = __builtin_amdgcn_exp2f(x * LOG2E_);
    float sp = LN2_ * __builtin_amdgcn_logf(1.0f + e);
    return (x > 20.0f) ? x : sp;   // v_cndmask, no exec-mask branch
}

__global__ __launch_bounds__(256)
void liquid_hier_kernel(const float* __restrict__ x,
                        const float* __restrict__ W_in_f,  const float* __restrict__ b_in_f,
                        const float* __restrict__ W_tau_f, const float* __restrict__ b_tau_f,
                        const float* __restrict__ W_in_m,  const float* __restrict__ b_in_m,
                        const float* __restrict__ W_tau_m, const float* __restrict__ b_tau_m,
                        const float* __restrict__ W_in_s,  const float* __restrict__ b_in_s,
                        const float* __restrict__ W_tau_s, const float* __restrict__ b_tau_s,
                        const float* __restrict__ W_h1,    const float* __restrict__ b_h1,
                        const float* __restrict__ W_h2,    const float* __restrict__ b_h2,
                        float* __restrict__ out)
{
    __shared__ _Float16 h16[16][64];       // hidden state (f16) = WMMA A source
    __shared__ float    taubuf[16][64];    // tau gate exchange (tau waves -> g waves)
    __shared__ float    xs[TT_][16][IN_];  // staged x chunk
    __shared__ float    fastL[16][64];
    __shared__ float    medIn[16][64];
    __shared__ float    medL[16][64];
    __shared__ float    slowL[16][64];
    __shared__ float    headL[16][128];

    const int tid  = threadIdx.x;
    const int lane = tid & 31;
    const int w    = tid >> 5;          // wave 0..7
    const int lh   = (lane >> 4) & 1;   // half-wave 0/1
    const int n    = lane & 15;         // N within 16x16 tile
    const bool isTau = (w >= 4);
    const int colBase = (isTau ? (w - 4) : w) * 16;
    const int col = colBase + n;
    const int rowBase = blockIdx.x * 16;

    // ---- per-wave resident B fragments: W rows 5..68 (h-part), our 16-col slice ----
    // B order used here (contiguous k, same as A): lane l -> N = l%16, element e -> K = 16*(l/16)+e
    const float* Wg = isTau ? W_tau_f : W_in_f;
    const float* bg = isTau ? b_tau_f : b_in_f;
    v16h b0, b1;
    #pragma unroll
    for (int e = 0; e < 16; ++e) {
        int K = 16 * lh + e;
        b0[e] = (_Float16)Wg[(5 + K)      * H_ + col];
        b1[e] = (_Float16)Wg[(5 + 32 + K) * H_ + col];
    }
    float wx[IN_];
    #pragma unroll
    for (int i = 0; i < IN_; ++i) wx[i] = Wg[i * H_ + col];
    const float bias = bg[col];

    // zero h state
    for (int idx = tid; idx < 16 * 64; idx += 256)
        h16[idx >> 6][idx & 63] = (_Float16)0.0f;
    float hreg[8], sreg[8];
    #pragma unroll
    for (int v = 0; v < 8; ++v) { hreg[v] = 0.0f; sreg[v] = 0.0f; }
    __syncthreads();

    for (int t0 = 0; t0 < T_; t0 += TT_) {
        // ---- stage x[rowBase..+15, t0..t0+TT) into LDS (160 contiguous floats/row) ----
        for (int idx = tid; idx < 16 * TT_ * IN_; idx += 256) {
            int r   = idx / (TT_ * IN_);
            int f   = idx % (TT_ * IN_);
            int ttx = f / IN_;
            int i   = f % IN_;
            size_t gaddr = ((size_t)(rowBase + r) * T_ + (size_t)(t0 + ttx)) * IN_ + i;
            xs[ttx][r][i] = x[gaddr];
            if ((t0 + TT_ < T_) && ((f & 15) == 0))
                __builtin_prefetch(&x[gaddr + (size_t)TT_ * IN_], 0, 0);  // global_prefetch
        }
        __syncthreads();

        for (int tt = 0; tt < TT_; ++tt) {
            // ---- C init: bias + x_t . Wx   (C layout: VGPR v -> M = v + 8*lh, N = col) ----
            v8f acc;
            #pragma unroll
            for (int v = 0; v < 8; ++v) {
                int m = v + 8 * lh;
                float s = bias;
                #pragma unroll
                for (int i = 0; i < IN_; ++i) s += xs[tt][m][i] * wx[i];
                acc[v] = s;
            }

            // ---- A fragments from LDS h16 (contiguous k order, same as B): lane l -> M = l%16,
            //      elems 0..15 = K[16*lh .. +15] (+32 for chunk 1)
            const _Float16* hrow = &h16[n][0];
            const v8h a0lo = *(const v8h*)(hrow +      16 * lh);
            const v8h a0hi = *(const v8h*)(hrow +  8 + 16 * lh);
            const v8h a1lo = *(const v8h*)(hrow + 32 + 16 * lh);
            const v8h a1hi = *(const v8h*)(hrow + 40 + 16 * lh);
            v16h a0, a1;
            #pragma unroll
            for (int i = 0; i < 8; ++i) {
                a0[i] = a0lo[i]; a0[i + 8] = a0hi[i];
                a1[i] = a1lo[i]; a1[i + 8] = a1hi[i];
            }

            acc = __builtin_amdgcn_wmma_f32_16x16x32_f16(false, a0, false, b0,
                                                         (short)0, acc, false, false);
            acc = __builtin_amdgcn_wmma_f32_16x16x32_f16(false, a1, false, b1,
                                                         (short)0, acc, false, false);

            if (isTau) {
                #pragma unroll
                for (int v = 0; v < 8; ++v)
                    taubuf[v + 8 * lh][col] = softplus_f(acc[v]) + 0.001f;
            }
            __syncthreads();
            if (!isTau) {
                #pragma unroll
                for (int v = 0; v < 8; ++v) {
                    int r = v + 8 * lh;
                    float gval = fast_tanh(acc[v]);
                    float tau  = taubuf[r][col];
                    float hn   = hreg[v] + (gval - hreg[v]) * __builtin_amdgcn_rcpf(tau);
                    hreg[v] = hn;
                    sreg[v] += hn;
                    h16[r][col] = (_Float16)hn;
                }
            }
            __syncthreads();
        }
    }

    // ---- epilogue: fast_last + mean -> medium -> slow -> head -> out ----
    if (!isTau) {
        #pragma unroll
        for (int v = 0; v < 8; ++v) {
            int r = v + 8 * lh;
            fastL[r][col] = hreg[v];
            medIn[r][col] = sreg[v] * (1.0f / (float)T_);
        }
    }
    __syncthreads();

    // medium cell, h0 = 0: med = tanh(p_in) / (softplus(p_tau)+1e-3); only rows 0..63 of W matter
    for (int idx = tid; idx < 16 * 64; idx += 256) {
        int r = idx >> 6, c = idx & 63;
        float pin = b_in_m[c], pt = b_tau_m[c];
        for (int k = 0; k < 64; ++k) {
            float mv = medIn[r][k];
            pin += mv * W_in_m[k * 64 + c];
            pt  += mv * W_tau_m[k * 64 + c];
        }
        float tau = softplus_f(pt) + 0.001f;
        medL[r][c] = fast_tanh(pin) * __builtin_amdgcn_rcpf(tau);
    }
    __syncthreads();

    for (int idx = tid; idx < 16 * 64; idx += 256) {
        int r = idx >> 6, c = idx & 63;
        float pin = b_in_s[c], pt = b_tau_s[c];
        for (int k = 0; k < 64; ++k) {
            float mv = medL[r][k];
            pin += mv * W_in_s[k * 64 + c];
            pt  += mv * W_tau_s[k * 64 + c];
        }
        float tau = softplus_f(pt) + 0.001f;
        slowL[r][c] = fast_tanh(pin) * __builtin_amdgcn_rcpf(tau);
    }
    __syncthreads();

    for (int idx = tid; idx < 16 * 128; idx += 256) {
        int r = idx >> 7, j = idx & 127;
        float s = b_h1[j];
        for (int k = 0; k < 64; ++k) s += fastL[r][k] * W_h1[k * 128 + j];
        for (int k = 0; k < 64; ++k) s += medL[r][k]  * W_h1[(64 + k) * 128 + j];
        for (int k = 0; k < 64; ++k) s += slowL[r][k] * W_h1[(128 + k) * 128 + j];
        headL[r][j] = fmaxf(s, 0.0f);
    }
    __syncthreads();

    for (int idx = tid; idx < 16 * 3; idx += 256) {
        int r = idx / 3, o = idx % 3;
        float s = b_h2[o];
        for (int k = 0; k < 128; ++k) s += headL[r][k] * W_h2[k * 3 + o];
        out[(size_t)(rowBase + r) * 3 + o] = s;
    }
}

extern "C" void kernel_launch(void* const* d_in, const int* in_sizes, int n_in,
                              void* d_out, int out_size, void* d_ws, size_t ws_size,
                              hipStream_t stream) {
    (void)in_sizes; (void)n_in; (void)out_size; (void)d_ws; (void)ws_size;
    const float* x       = (const float*)d_in[0];
    const float* W_in_f  = (const float*)d_in[1];
    const float* b_in_f  = (const float*)d_in[2];
    const float* W_tau_f = (const float*)d_in[3];
    const float* b_tau_f = (const float*)d_in[4];
    const float* W_in_m  = (const float*)d_in[5];
    const float* b_in_m  = (const float*)d_in[6];
    const float* W_tau_m = (const float*)d_in[7];
    const float* b_tau_m = (const float*)d_in[8];
    const float* W_in_s  = (const float*)d_in[9];
    const float* b_in_s  = (const float*)d_in[10];
    const float* W_tau_s = (const float*)d_in[11];
    const float* b_tau_s = (const float*)d_in[12];
    const float* W_h1    = (const float*)d_in[13];
    const float* b_h1    = (const float*)d_in[14];
    const float* W_h2    = (const float*)d_in[15];
    const float* b_h2    = (const float*)d_in[16];
    float* out = (float*)d_out;

    dim3 grid(B_ / 16);
    dim3 block(256);
    liquid_hier_kernel<<<grid, block, 0, stream>>>(x,
        W_in_f, b_in_f, W_tau_f, b_tau_f,
        W_in_m, b_in_m, W_tau_m, b_tau_m,
        W_in_s, b_in_s, W_tau_s, b_tau_s,
        W_h1, b_h1, W_h2, b_h2, out);
}
